// BasicBlock_14654428414031
// MI455X (gfx1250) — matrix operand order repaired and checked by reading, not compiled, not run on hardware
//
#include <hip/hip_runtime.h>

#define NB   32
#define CC   128
#define HH   56
#define WWW  56
#define HP   58                  // padded spatial
#define HW   (HH*WWW)            // 3136
#define HWP  (HP*HP)             // 3364
#define PIX  (NB*HW)             // 100352
#define PIXP (NB*HWP)            // 107648
#define BNEPS 1e-5f

// scheduling-group pipelining (mask: 0x20 = VMEM read, 0x8 = MFMA/WMMA)
#if __has_builtin(__builtin_amdgcn_sched_group_barrier)
#define SGB(mask, size) __builtin_amdgcn_sched_group_barrier((mask), (size), 0)
#else
#define SGB(mask, size)
#endif

typedef __attribute__((ext_vector_type(16))) __bf16 v16bf;
typedef __attribute__((ext_vector_type(8)))  __bf16 v8bf;
typedef __attribute__((ext_vector_type(8)))  float  v8f;

__device__ __forceinline__ __bf16 f2bf(float f) {
    unsigned u = __builtin_bit_cast(unsigned, f);
    unsigned r = (u + 0x7FFFu + ((u >> 16) & 1u)) >> 16;
    unsigned short s = (unsigned short)r;
    return __builtin_bit_cast(__bf16, s);
}

// ---------------- zero stats ----------------
__global__ void k_zero(float* p, int n) {
    int i = blockIdx.x * blockDim.x + threadIdx.x;
    if (i < n) p[i] = 0.f;
}

// ------- x fp32 NCHW -> bf16 zero-padded [N][58][58][C] -------
__global__ void k_pack_x(const float* __restrict__ x, __bf16* __restrict__ xb) {
    int e = blockIdx.x * 256 + threadIdx.x;      // e < PIXP*CC (exact grid)
    int c = e & (CC - 1);
    int q = e >> 7;
    int n = q / HWP;
    int r = q - n * HWP;
    int hh = r / HP;
    int ww = r - hh * HP;
    int h = hh - 1, w = ww - 1;
    float v = 0.f;
    if ((unsigned)h < (unsigned)HH && (unsigned)w < (unsigned)WWW)
        v = x[((n * CC + c) * HH + h) * WWW + w];
    xb[e] = f2bf(v);
}

// ------- sign(w) OIHW -> bf16 packed in WMMA-B fragment layout -------
// flat e = (((tap*4 + cic)*8 + nf)*32 + lane)*16 + i
// lane holds col n = lane&15 (cout), k = (i>>3)*16 + (lane>>4)*8 + (i&7) (ci within 32-chunk)
__global__ void k_pack_w(const float* __restrict__ w, __bf16* __restrict__ wp) {
    int e    = blockIdx.x * 256 + threadIdx.x;   // e < 9*4*8*32*16 = 147456
    int i    = e & 15;
    int lane = (e >> 4) & 31;
    int nf   = (e >> 9) & 7;
    int cic  = (e >> 12) & 3;
    int tap  = e >> 14;
    int co   = nf * 16 + (lane & 15);
    int ci   = cic * 32 + (i >> 3) * 16 + (lane >> 4) * 8 + (i & 7);
    float v  = w[(co * CC + ci) * 9 + tap];
    float s  = (v > 0.f) ? 1.f : ((v < 0.f) ? -1.f : 0.f);
    wp[e] = f2bf(s);
}

// ---------------- binarized 3x3 conv via WMMA implicit GEMM ----------------
// act: bf16 padded [N][58][58][C]; wp: packed B frags; y: fp32 dense NHWC.
// sched_group_barrier-pipelined: step s+1's 10 loads placed before step s's WMMAs.
__global__ void __launch_bounds__(256) k_conv(const __bf16* __restrict__ act,
                                              const __bf16* __restrict__ wp,
                                              float* __restrict__ y,
                                              float* __restrict__ gsum,
                                              float* __restrict__ gsq) {
    __shared__ float lsum[CC];
    __shared__ float lsq[CC];
    if (threadIdx.x < CC) { lsum[threadIdx.x] = 0.f; lsq[threadIdx.x] = 0.f; }
    __syncthreads();

    const int lane = threadIdx.x & 31;
    const int wid  = threadIdx.x >> 5;
    const int mi   = wid & 3;        // 16-pixel M fragment
    const int nh   = wid >> 2;       // 0/1: which 64-channel half

    // this lane's output pixel (A-matrix row m = lane&15)
    const int p = blockIdx.x * 64 + mi * 16 + (lane & 15);
    const int n = p / HW;
    const int r = p - n * HW;
    const int h = r / WWW;
    const int w = r - h * WWW;
    const int klo = (lane >> 4) * 8;   // A layout: lanes 16-31 hold k+8 runs
    // padded-center base for this pixel; all tap/cic offsets are immediates
    const __bf16* abase = act + (size_t)((n * HP + (h + 1)) * HP + (w + 1)) * CC + klo;
    // loop-invariant per-lane B base; all frag offsets become immediates
    const __bf16* bbase = wp + nh * 4 * 512 + lane * 16;

    union AU { v16bf v; v8bf h8[2]; };
    AU    A[2];
    v16bf B[2][4];

    auto loadA = [&](int s, AU& dst) {
        const int cic = s / 9, tap = s % 9;
        const int off = ((tap / 3 - 1) * HP + (tap % 3 - 1)) * CC + cic * 32;
        dst.h8[0] = *(const v8bf*)(abase + off);
        dst.h8[1] = *(const v8bf*)(abase + off + 16);
    };
    auto loadB = [&](int s, v16bf* dst) {
        const int cic = s / 9, tap = s % 9;
        const __bf16* bp = bbase + (size_t)((tap * 4 + cic) * 8) * 512;
        dst[0] = *(const v16bf*)(bp);
        dst[1] = *(const v16bf*)(bp + 512);
        dst[2] = *(const v16bf*)(bp + 1024);
        dst[3] = *(const v16bf*)(bp + 1536);
    };

    v8f acc0 = {}, acc1 = {}, acc2 = {}, acc3 = {};

    loadA(0, A[0]);
    loadB(0, B[0]);
    SGB(0x020, 10);                       // step-0 loads first
#pragma unroll
    for (int s = 0; s < 36; ++s) {        // (cic, tap) flattened: 4 * 9
        const int cur = s & 1, nxt = cur ^ 1;
        if (s < 35) {                     // prefetch next step
            loadA(s + 1, A[nxt]);
            loadB(s + 1, B[nxt]);
        }
        acc0 = __builtin_amdgcn_wmma_f32_16x16x32_bf16(false, A[cur].v, false, B[cur][0], (short)0, acc0, false, false);
        acc1 = __builtin_amdgcn_wmma_f32_16x16x32_bf16(false, A[cur].v, false, B[cur][1], (short)0, acc1, false, false);
        acc2 = __builtin_amdgcn_wmma_f32_16x16x32_bf16(false, A[cur].v, false, B[cur][2], (short)0, acc2, false, false);
        acc3 = __builtin_amdgcn_wmma_f32_16x16x32_bf16(false, A[cur].v, false, B[cur][3], (short)0, acc3, false, false);
        if (s < 35) SGB(0x020, 10);       // step s+1's loads...
        SGB(0x008, 4);                    // ...before step s's 4 WMMAs
    }

    // Epilogue: C/D layout -> pixel m = v + (lane>>4)*8, channel = lane&15
    const int mbase = blockIdx.x * 64 + mi * 16 + (lane >> 4) * 8;
    const int cl = lane & 15;
    auto emit = [&](const v8f& a, int j) {
        const int co = nh * 64 + j * 16 + cl;
        float s = 0.f, q = 0.f;
#pragma unroll
        for (int v = 0; v < 8; ++v) {
            float val = a[v];
            s += val;
            q += val * val;
            y[(size_t)(mbase + v) * CC + co] = val;
        }
        // lanes l and l+16 own the same channel: fold halves, halve LDS atomics
        s += __shfl_xor(s, 16, 32);
        q += __shfl_xor(q, 16, 32);
        if (lane < 16) {
            atomicAdd(&lsum[co], s);   // ds_add_f32
            atomicAdd(&lsq[co], q);
        }
    };
    emit(acc0, 0); emit(acc1, 1); emit(acc2, 2); emit(acc3, 3);
    __syncthreads();
    if (threadIdx.x < CC) {
        atomicAdd(&gsum[threadIdx.x], lsum[threadIdx.x]);
        atomicAdd(&gsq[threadIdx.x], lsq[threadIdx.x]);
    }
}

// ------- BN1 (batch stats) + ReLU -> bf16 zero-padded [N][58][58][C] -------
__global__ void k_bn_relu_pack(const float* __restrict__ y,
                               const float* __restrict__ gsum, const float* __restrict__ gsq,
                               const float* __restrict__ gamma, const float* __restrict__ beta,
                               __bf16* __restrict__ a1) {
    int e = blockIdx.x * 256 + threadIdx.x;      // over padded domain
    int c = e & (CC - 1);
    int q = e >> 7;
    int n = q / HWP;
    int r = q - n * HWP;
    int hh = r / HP;
    int ww = r - hh * HP;
    int h = hh - 1, w = ww - 1;
    float v = 0.f;
    if ((unsigned)h < (unsigned)HH && (unsigned)w < (unsigned)WWW) {
        int p = (n * HH + h) * WWW + w;
        float cnt  = (float)PIX;
        float mean = gsum[c] / cnt;
        float var  = gsq[c] / cnt - mean * mean;
        float inv  = rsqrtf(var + BNEPS);
        v = (y[(size_t)p * CC + c] - mean) * inv * gamma[c] + beta[c];
        v = v > 0.f ? v : 0.f;
    }
    a1[e] = f2bf(v);
}

// ---------------- BN2 + residual + ReLU -> fp32 NCHW ----------------
__global__ void k_bn_add_relu(const float* __restrict__ y,
                              const float* __restrict__ gsum, const float* __restrict__ gsq,
                              const float* __restrict__ gamma, const float* __restrict__ beta,
                              const float* __restrict__ x, float* __restrict__ out) {
    int e = blockIdx.x * 256 + threadIdx.x;      // NCHW flat
    int w = e % WWW;
    int t = e / WWW;
    int h = t % HH;
    t /= HH;
    int c = t % CC;
    int n = t / CC;
    int p = (n * HH + h) * WWW + w;
    float cnt  = (float)PIX;
    float mean = gsum[c] / cnt;
    float var  = gsq[c] / cnt - mean * mean;
    float inv  = rsqrtf(var + BNEPS);
    float v = (y[(size_t)p * CC + c] - mean) * inv * gamma[c] + beta[c] + x[e];
    out[e] = v > 0.f ? v : 0.f;
}

extern "C" void kernel_launch(void* const* d_in, const int* in_sizes, int n_in,
                              void* d_out, int out_size, void* d_ws, size_t ws_size,
                              hipStream_t stream) {
    const float* x  = (const float*)d_in[0];
    const float* w1 = (const float*)d_in[1];
    const float* g1 = (const float*)d_in[2];
    const float* b1 = (const float*)d_in[3];
    const float* w2 = (const float*)d_in[4];
    const float* g2 = (const float*)d_in[5];
    const float* b2 = (const float*)d_in[6];
    float* out = (float*)d_out;

    char* ws = (char*)d_ws;
    const size_t XBP_BYTES = (size_t)PIXP * CC * 2;    // 27,557,888 (padded bf16)
    const size_t Y_BYTES   = (size_t)PIX * CC * 4;     // 51,380,224
    const size_t WP_BYTES  = 9 * 4 * 8 * 32 * 16 * 2;  // 294,912
    __bf16* xb  = (__bf16*)ws;                         // reused as a1 (padded)
    float*  y   = (float*)(ws + XBP_BYTES);
    __bf16* wp1 = (__bf16*)(ws + XBP_BYTES + Y_BYTES);
    __bf16* wp2 = (__bf16*)(ws + XBP_BYTES + Y_BYTES + WP_BYTES);
    float*  st  = (float*)(ws + XBP_BYTES + Y_BYTES + 2 * WP_BYTES);
    float* s1 = st, *q1 = st + 128, *s2 = st + 256, *q2 = st + 384;

    k_zero<<<2, 256, 0, stream>>>(st, 512);
    k_pack_x<<<PIXP * CC / 256, 256, 0, stream>>>(x, xb);
    k_pack_w<<<576, 256, 0, stream>>>(w1, wp1);
    k_pack_w<<<576, 256, 0, stream>>>(w2, wp2);
    k_conv<<<PIX / 64, 256, 0, stream>>>(xb, wp1, y, s1, q1);
    k_bn_relu_pack<<<PIXP * CC / 256, 256, 0, stream>>>(y, s1, q1, g1, b1, xb);
    k_conv<<<PIX / 64, 256, 0, stream>>>(xb, wp2, y, s2, q2);
    k_bn_add_relu<<<PIX * CC / 256, 256, 0, stream>>>(y, s2, q2, g2, b2, x, out);
}
